// BitDecoderBlock_37125697307133
// MI455X (gfx1250) — 16-bit and 8-bit matrix operand order repaired and checked by reading, compiled, not run on hardware
//
#include <hip/hip_runtime.h>
#include <hip/hip_fp16.h>
#include <stdint.h>

typedef int      v8i  __attribute__((ext_vector_type(8)));
typedef float    v8f  __attribute__((ext_vector_type(8)));
typedef _Float16 v16h __attribute__((ext_vector_type(16)));

constexpr int Bx = 2, Tx = 2048, Cx = 2048, FFx = 8192, Hx = 16, Dx = 128;
constexpr int BT = Bx * Tx;

__global__ __launch_bounds__(256)
void absmean_partial(const float* __restrict__ w, size_t n, float* __restrict__ partial)
{
    float s = 0.f;
    for (size_t i = (size_t)blockIdx.x * 256 + threadIdx.x; i < n; i += (size_t)gridDim.x * 256)
        s += fabsf(w[i]);
    __shared__ float sb[256];
    sb[threadIdx.x] = s;
    __syncthreads();
    for (int k = 128; k > 0; k >>= 1) {
        if ((int)threadIdx.x < k) sb[threadIdx.x] += sb[threadIdx.x + k];
        __syncthreads();
    }
    if (threadIdx.x == 0) partial[blockIdx.x] = sb[0];
}

__global__ __launch_bounds__(256)
void wscale_finalize(const float* __restrict__ partial, float* __restrict__ sc)
{
    __shared__ float sb[256];
    for (int m = 0; m < 7; ++m) {
        float s = 0.f;
        for (int j = threadIdx.x; j < 1024; j += 256) s += partial[m * 1024 + j];
        sb[threadIdx.x] = s;
        __syncthreads();
        for (int k = 128; k > 0; k >>= 1) {
            if ((int)threadIdx.x < k) sb[threadIdx.x] += sb[threadIdx.x + k];
            __syncthreads();
        }
        if (threadIdx.x == 0) {
            const float cnt  = (m < 4) ? (float)((size_t)Cx * Cx) : (float)((size_t)FFx * Cx);
            const float mean = sb[0] / cnt;
            const float im   = fmaxf(mean, 1e-5f);
            sc[m]     = im;
            sc[8 + m] = 1.0f / im;
        }
        __syncthreads();
    }
}

__global__ __launch_bounds__(256)
void quant_weight(const float* __restrict__ w, int8_t* __restrict__ wq, size_t n,
                  const float* __restrict__ sc, int idx)
{
    const float qs = sc[8 + idx];
    for (size_t i = (size_t)blockIdx.x * 256 + threadIdx.x; i < n; i += (size_t)gridDim.x * 256) {
        const float t = fminf(fmaxf(rintf(w[i] * qs), -1.f), 1.f);
        wq[i] = (int8_t)t;
    }
}

template <bool PRE, typename TIN>
__global__ __launch_bounds__(256)
void act_quant(const TIN* __restrict__ x, const float* __restrict__ nw, int W,
               int8_t* __restrict__ q, float* __restrict__ inv_as)
{
    const int row = blockIdx.x;
    const int tid = threadIdx.x;
    const TIN* xr = x + (size_t)row * W;

    float s1 = 0.f, s2 = 0.f, am = 0.f;
    for (int j = tid; j < W; j += 256) {
        const float v  = (float)xr[j];
        const float wv = nw ? nw[j] : 1.0f;
        const float vw = v * wv;
        s1 += v * v;
        s2 += vw * vw;
        am  = fmaxf(am, fabsf(vw));
    }
    __shared__ float r1[256], r2[256], r3[256];
    r1[tid] = s1; r2[tid] = s2; r3[tid] = am;
    __syncthreads();
    for (int k = 128; k > 0; k >>= 1) {
        if (tid < k) {
            r1[tid] += r1[tid + k];
            r2[tid] += r2[tid + k];
            r3[tid]  = fmaxf(r3[tid], r3[tid + k]);
        }
        __syncthreads();
    }
    __shared__ float bc;
    if (tid == 0) {
        const float iw  = 1.0f / (float)W;
        const float g1  = PRE ? rsqrtf(r1[0] * iw + 1e-6f) : 1.0f;
        const float g2  = rsqrtf(g1 * g1 * r2[0] * iw + 1e-6f);
        const float g   = g1 * g2;
        const float amx = fmaxf(r3[0] * g, 1e-5f);
        bc = g * (127.0f / amx);
        inv_as[row] = amx * (1.0f / 127.0f);
    }
    __syncthreads();
    const float f = bc;
    for (int j = tid; j < W; j += 256) {
        const float v  = (float)xr[j];
        const float wv = nw ? nw[j] : 1.0f;
        const float t  = fminf(fmaxf(rintf(v * wv * f), -128.0f), 127.0f);
        q[(size_t)row * W + j] = (int8_t)t;
    }
}

enum { EPI_RES = 0, EPI_H16 = 1, EPI_QKL = 2, EPI_VTL = 3 };

template <int EPI>
__global__ __launch_bounds__(128)
void gemm_i8(const int8_t* __restrict__ Aq, const float* __restrict__ ars,
             const int8_t* __restrict__ Wq, const float* __restrict__ sc, int scIdx,
             int M, int N, int K,
             const float* __restrict__ res, float* __restrict__ outf,
             _Float16* __restrict__ outh)
{
    __shared__ alignas(16) int8_t bsm[2][64 * 80];
    const int lane = threadIdx.x & 31;
    const int wvid = threadIdx.x >> 5;
    const int ln   = lane & 15;
    const int kh   = lane >> 4;
    const int mBase = blockIdx.y * 128;
    const int nBase = blockIdx.x * 64;

    const int8_t* arow0 = Aq + (size_t)(mBase + 32 * wvid + ln) * K;
    const int8_t* arow1 = arow0 + (size_t)16 * K;

    const int crow  = threadIdx.x >> 1;
    const int chalf = threadIdx.x & 1;
    const int8_t*  gsrc = Wq + (size_t)(nBase + crow) * K + chalf * 32;
    const uint32_t ldst = (uint32_t)(uintptr_t)&bsm[0][crow * 80 + chalf * 32];
    const uint32_t lstr = (uint32_t)(64 * 80);

    auto issue = [&](int kstep, int buf) {
        const unsigned long long g = (unsigned long long)(uintptr_t)(gsrc + (size_t)kstep * 64);
        const uint32_t l = ldst + (uint32_t)buf * lstr;
        asm volatile("global_load_async_to_lds_b128 %0, %1, off"
                     :: "v"(l), "v"(g) : "memory");
        asm volatile("global_load_async_to_lds_b128 %0, %1, off offset:16"
                     :: "v"(l), "v"(g) : "memory");
    };

    v8i acc[2][4];
#pragma unroll
    for (int r = 0; r < 2; ++r)
#pragma unroll
        for (int t = 0; t < 4; ++t)
#pragma unroll
            for (int i = 0; i < 8; ++i) acc[r][t][i] = 0;

    const int S = K >> 6;
    issue(0, 0);
    for (int s = 0; s < S; ++s) {
        const int k0  = s << 6;
        const int buf = s & 1;
        if (s + 1 < S) {
            issue(s + 1, buf ^ 1);
            asm volatile("s_wait_asynccnt 2" ::: "memory");
        } else {
            asm volatile("s_wait_asynccnt 0" ::: "memory");
        }
        __syncthreads();

        union { unsigned long long u[4]; v8i v; } a0, a1;
        const int8_t* ap0 = arow0 + k0 + 16 * kh;
        const int8_t* ap1 = arow1 + k0 + 16 * kh;
        a0.u[0] = *(const unsigned long long*)(ap0);
        a0.u[1] = *(const unsigned long long*)(ap0 + 8);
        a0.u[2] = *(const unsigned long long*)(ap0 + 32);
        a0.u[3] = *(const unsigned long long*)(ap0 + 40);
        a1.u[0] = *(const unsigned long long*)(ap1);
        a1.u[1] = *(const unsigned long long*)(ap1 + 8);
        a1.u[2] = *(const unsigned long long*)(ap1 + 32);
        a1.u[3] = *(const unsigned long long*)(ap1 + 40);
        if (s + 1 < S) {
            __builtin_prefetch(ap0 + 64, 0, 1);
            __builtin_prefetch(ap1 + 64, 0, 1);
        }
#pragma unroll
        for (int t = 0; t < 4; ++t) {
            union { uint4 u[2]; v8i v; } b;
            const int8_t* bp = &bsm[buf][(16 * t + ln) * 80 + 16 * kh];
            b.u[0] = *(const uint4*)(bp);
            b.u[1] = *(const uint4*)(bp + 32);
            acc[0][t] = __builtin_amdgcn_wmma_i32_16x16x64_iu8(true, a0.v, true, b.v,
                                                               acc[0][t], false, false);
            acc[1][t] = __builtin_amdgcn_wmma_i32_16x16x64_iu8(true, a1.v, true, b.v,
                                                               acc[1][t], false, false);
        }
        __syncthreads();
    }

    const float wsc = sc[scIdx];
#pragma unroll
    for (int r = 0; r < 2; ++r) {
        const int m0 = mBase + 32 * wvid + 16 * r + 8 * kh;
        float ias[8];
#pragma unroll
        for (int i = 0; i < 8; ++i) ias[i] = ars[m0 + i] * wsc;
#pragma unroll
        for (int t = 0; t < 4; ++t) {
            const int n = nBase + 16 * t + ln;
#pragma unroll
            for (int i = 0; i < 8; ++i) {
                const int   m = m0 + i;
                const float f = (float)acc[r][t][i] * ias[i];
                if constexpr (EPI == EPI_RES) {
                    outf[(size_t)m * N + n] = res[(size_t)m * N + n] + f;
                } else if constexpr (EPI == EPI_H16) {
                    outh[(size_t)m * N + n] = (_Float16)f;
                } else {
                    const int bb = m >> 11, tt = m & (Tx - 1);
                    const int hh = n >> 7,  dd = n & (Dx - 1);
                    if constexpr (EPI == EPI_QKL)
                        outh[(((size_t)(bb * Hx + hh)) * Tx + tt) * Dx + dd] = (_Float16)f;
                    else
                        outh[(((size_t)(bb * Hx + hh)) * Dx + dd) * Tx + tt] = (_Float16)f;
                }
            }
        }
    }
}

__global__ __launch_bounds__(256)
void rope_k(_Float16* __restrict__ buf, int npairs)
{
    const int idx = blockIdx.x * 256 + threadIdx.x;
    if (idx >= npairs) return;
    const int    d2   = idx & 63;
    const int    t    = (idx >> 6) & (Tx - 1);
    const size_t base = ((size_t)(idx >> 6)) * Dx + 2 * d2;
    const float  fr   = __powf(10000.0f, -(float)d2 * (1.0f / 64.0f));
    const float  ang  = (float)t * fr;
    float sn, cs;
    __sincosf(ang, &sn, &cs);
    const float xr = (float)buf[base];
    const float xi = (float)buf[base + 1];
    buf[base]     = (_Float16)(xr * cs - xi * sn);
    buf[base + 1] = (_Float16)(xr * sn + xi * cs);
}

__global__ __launch_bounds__(128)
void attn_fa(const _Float16* __restrict__ qf, const _Float16* __restrict__ kf,
             const _Float16* __restrict__ vt, float* __restrict__ y)
{
    __shared__ alignas(16) _Float16 pls[4][16][32];
    const int lane = threadIdx.x & 31;
    const int wvid = threadIdx.x >> 5;
    const int ln   = lane & 15;
    const int kh   = lane >> 4;
    const int tile = blockIdx.x * 4 + wvid;
    const int b  = tile >> 11;
    const int h  = (tile >> 7) & (Hx - 1);
    const int q0 = (tile & 127) << 4;
    const size_t bh = (size_t)(b * Hx + h);

    v16h qa[4];
    {
        const _Float16* qrow = qf + (bh * Tx + (q0 + ln)) * Dx;
#pragma unroll
        for (int c = 0; c < 4; ++c) {
            union { uint4 u[2]; v16h v; } t;
            t.u[0] = *(const uint4*)(qrow + 32 * c + 16 * kh);
            t.u[1] = *(const uint4*)(qrow + 32 * c + 16 * kh + 8);
            qa[c] = t.v;
        }
    }

    v8f  O[8];
    float mrun[8], lrun[8];
#pragma unroll
    for (int j = 0; j < 8; ++j)
#pragma unroll
        for (int i = 0; i < 8; ++i) O[j][i] = 0.f;
#pragma unroll
    for (int i = 0; i < 8; ++i) { mrun[i] = -3.0e38f; lrun[i] = 0.f; }

    const float SC = 0.08838834764831845f;
    const int nsteps = (q0 + 47) >> 5;
    for (int st = 0; st < nsteps; ++st) {
        const int ks = st * 32;
        v8f s0, s1;
#pragma unroll
        for (int i = 0; i < 8; ++i) { s0[i] = 0.f; s1[i] = 0.f; }
        {
            const _Float16* k0r = kf + (bh * Tx + (ks + ln)) * Dx;
            const _Float16* k1r = k0r + 16 * Dx;
#pragma unroll
            for (int c = 0; c < 4; ++c) {
                union { uint4 u[2]; v16h v; } kb;
                kb.u[0] = *(const uint4*)(k0r + 32 * c + 16 * kh);
                kb.u[1] = *(const uint4*)(k0r + 32 * c + 16 * kh + 8);
                s0 = __builtin_amdgcn_wmma_f32_16x16x32_f16(false, qa[c], false, kb.v,
                                                            (short)0, s0, false, false);
            }
#pragma unroll
            for (int c = 0; c < 4; ++c) {
                union { uint4 u[2]; v16h v; } kb;
                kb.u[0] = *(const uint4*)(k1r + 32 * c + 16 * kh);
                kb.u[1] = *(const uint4*)(k1r + 32 * c + 16 * kh + 8);
                s1 = __builtin_amdgcn_wmma_f32_16x16x32_f16(false, qa[c], false, kb.v,
                                                            (short)0, s1, false, false);
            }
        }
        float corr[8];
#pragma unroll
        for (int i = 0; i < 8; ++i) {
            const int   r  = q0 + 8 * kh + i;
            const float a0 = (ks + ln      <= r) ? s0[i] * SC : -3.0e38f;
            const float a1 = (ks + 16 + ln <= r) ? s1[i] * SC : -3.0e38f;
            float tm = fmaxf(a0, a1);
            tm = fmaxf(tm, __shfl_xor(tm, 1));
            tm = fmaxf(tm, __shfl_xor(tm, 2));
            tm = fmaxf(tm, __shfl_xor(tm, 4));
            tm = fmaxf(tm, __shfl_xor(tm, 8));
            const float mn = fmaxf(mrun[i], tm);
            const float p0 = __expf(a0 - mn);
            const float p1 = __expf(a1 - mn);
            float rs = p0 + p1;
            rs += __shfl_xor(rs, 1);
            rs += __shfl_xor(rs, 2);
            rs += __shfl_xor(rs, 4);
            rs += __shfl_xor(rs, 8);
            corr[i] = __expf(mrun[i] - mn);
            lrun[i] = lrun[i] * corr[i] + rs;
            mrun[i] = mn;
            pls[wvid][8 * kh + i][ln]      = (_Float16)p0;
            pls[wvid][8 * kh + i][16 + ln] = (_Float16)p1;
        }
#pragma unroll
        for (int j = 0; j < 8; ++j)
#pragma unroll
            for (int i = 0; i < 8; ++i) O[j][i] *= corr[i];

        union { uint4 u[2]; v16h v; } pa;
        pa.u[0] = *(const uint4*)&pls[wvid][ln][16 * kh];
        pa.u[1] = *(const uint4*)&pls[wvid][ln][16 * kh + 8];
#pragma unroll
        for (int j = 0; j < 8; ++j) {
            const _Float16* vrow = vt + (bh * Dx + 16 * j + ln) * Tx;
            union { uint4 u[2]; v16h v; } vb;
            vb.u[0] = *(const uint4*)(vrow + ks + 16 * kh);
            vb.u[1] = *(const uint4*)(vrow + ks + 16 * kh + 8);
            O[j] = __builtin_amdgcn_wmma_f32_16x16x32_f16(false, pa.v, false, vb.v,
                                                          (short)0, O[j], false, false);
        }
    }

#pragma unroll
    for (int j = 0; j < 8; ++j)
#pragma unroll
        for (int i = 0; i < 8; ++i) {
            const int t = q0 + 8 * kh + i;
            y[((size_t)(b * Tx) + t) * Cx + h * Dx + 16 * j + ln] = O[j][i] / lrun[i];
        }
}

__global__ __launch_bounds__(256)
void silu_mul(_Float16* __restrict__ u1, const _Float16* __restrict__ u2, size_t n)
{
    for (size_t i = (size_t)blockIdx.x * 256 + threadIdx.x; i < n; i += (size_t)gridDim.x * 256) {
        const float a = (float)u1[i];
        const float s = a / (1.0f + __expf(-a));
        u1[i] = (_Float16)(s * (float)u2[i]);
    }
}

extern "C" void kernel_launch(void* const* d_in, const int* in_sizes, int n_in,
                              void* d_out, int out_size, void* d_ws, size_t ws_size,
                              hipStream_t stream)
{
    (void)in_sizes; (void)n_in; (void)out_size; (void)ws_size;

    const float* x   = (const float*)d_in[0];
    const float* wq  = (const float*)d_in[1];
    const float* wk  = (const float*)d_in[2];
    const float* wv  = (const float*)d_in[3];
    const float* wpj = (const float*)d_in[4];
    const float* w1  = (const float*)d_in[5];
    const float* w2  = (const float*)d_in[6];
    const float* w3  = (const float*)d_in[7];
    const float* anw = (const float*)d_in[8];
    const float* fnw = (const float*)d_in[9];

    char* p = (char*)d_ws;
    auto take = [&](size_t bytes) -> char* {
        char* r = p;
        p += (bytes + 255) & ~(size_t)255;
        return r;
    };

    const size_t CC = (size_t)Cx * Cx, FC = (size_t)FFx * Cx;

    float*    scales  = (float*)take(64 * sizeof(float));
    float*    partial = (float*)take(7 * 1024 * sizeof(float));
    int8_t*   q_wq    = (int8_t*)take(CC);
    int8_t*   q_wk    = (int8_t*)take(CC);
    int8_t*   q_wv    = (int8_t*)take(CC);
    int8_t*   q_wpj   = (int8_t*)take(CC);
    int8_t*   q_w1    = (int8_t*)take(FC);
    int8_t*   q_w2    = (int8_t*)take(FC);
    int8_t*   q_w3    = (int8_t*)take(FC);
    int8_t*   xq8     = (int8_t*)take((size_t)BT * Cx);
    int8_t*   uq8     = (int8_t*)take((size_t)BT * FFx);
    float*    invas   = (float*)take((size_t)BT * sizeof(float));
    _Float16* qh      = (_Float16*)take((size_t)BT * Cx * 2);
    _Float16* khb     = (_Float16*)take((size_t)BT * Cx * 2);
    _Float16* vth     = (_Float16*)take((size_t)BT * Cx * 2);
    float*    ybuf    = (float*)take((size_t)BT * Cx * 4);
    float*    x1      = (float*)take((size_t)BT * Cx * 4);
    _Float16* u1      = (_Float16*)take((size_t)BT * FFx * 2);
    _Float16* u2      = (_Float16*)take((size_t)BT * FFx * 2);

    const float* wsrc[7] = {wq, wk, wv, wpj, w1, w2, w3};
    int8_t*      wdst[7] = {q_wq, q_wk, q_wv, q_wpj, q_w1, q_w2, q_w3};
    const size_t wn[7]   = {CC, CC, CC, CC, FC, FC, FC};
    for (int m = 0; m < 7; ++m)
        absmean_partial<<<1024, 256, 0, stream>>>(wsrc[m], wn[m], partial + m * 1024);
    wscale_finalize<<<1, 256, 0, stream>>>(partial, scales);
    for (int m = 0; m < 7; ++m)
        quant_weight<<<4096, 256, 0, stream>>>(wsrc[m], wdst[m], wn[m], scales, m);

    const dim3 blk(128);
    const dim3 gC(Cx / 64, BT / 128);
    const dim3 gF(FFx / 64, BT / 128);

    act_quant<true, float><<<BT, 256, 0, stream>>>(x, anw, Cx, xq8, invas);
    gemm_i8<EPI_QKL><<<gC, blk, 0, stream>>>(xq8, invas, q_wq, scales, 0, BT, Cx, Cx,
                                             nullptr, nullptr, qh);
    gemm_i8<EPI_QKL><<<gC, blk, 0, stream>>>(xq8, invas, q_wk, scales, 1, BT, Cx, Cx,
                                             nullptr, nullptr, khb);
    gemm_i8<EPI_VTL><<<gC, blk, 0, stream>>>(xq8, invas, q_wv, scales, 2, BT, Cx, Cx,
                                             nullptr, nullptr, vth);
    const int npairs = BT * Cx / 2;
    rope_k<<<npairs / 256, 256, 0, stream>>>(qh, npairs);
    rope_k<<<npairs / 256, 256, 0, stream>>>(khb, npairs);
    attn_fa<<<(Bx * Hx * (Tx / 16)) / 4, blk, 0, stream>>>(qh, khb, vth, ybuf);

    act_quant<false, float><<<BT, 256, 0, stream>>>(ybuf, nullptr, Cx, xq8, invas);
    gemm_i8<EPI_RES><<<gC, blk, 0, stream>>>(xq8, invas, q_wpj, scales, 3, BT, Cx, Cx,
                                             x, x1, nullptr);

    act_quant<true, float><<<BT, 256, 0, stream>>>(x1, fnw, Cx, xq8, invas);
    gemm_i8<EPI_H16><<<gF, blk, 0, stream>>>(xq8, invas, q_w1, scales, 4, BT, FFx, Cx,
                                             nullptr, nullptr, u1);
    gemm_i8<EPI_H16><<<gF, blk, 0, stream>>>(xq8, invas, q_w2, scales, 5, BT, FFx, Cx,
                                             nullptr, nullptr, u2);
    silu_mul<<<8192, 256, 0, stream>>>(u1, u2, (size_t)BT * FFx);
    act_quant<false, _Float16><<<BT, 256, 0, stream>>>(u1, nullptr, FFx, uq8, invas);
    gemm_i8<EPI_RES><<<gC, blk, 0, stream>>>(uq8, invas, q_w3, scales, 6, BT, Cx, FFx,
                                             x1, (float*)d_out, nullptr);
}
